// GraphConvLayer_25116968747539
// MI455X (gfx1250) — matrix operand order repaired and checked by reading, not compiled, not run on hardware
//
#include <hip/hip_runtime.h>
#include <math.h>

// ---------------------------------------------------------------------------
// Problem constants (from reference)
// ---------------------------------------------------------------------------
static constexpr int B_   = 16;
static constexpr int S_   = 512;
static constexpr int D_   = 512;
static constexpr int DOM_ = 8;
static constexpr int L_   = 3;
static constexpr long long BSD = (long long)B_ * S_ * D_;   // 4,194,304
static constexpr long long SS  = (long long)S_ * S_;        // 262,144
static constexpr long long SD  = (long long)S_ * D_;        // 262,144

// ---------------------------------------------------------------------------
// CDNA5 WMMA types
// ---------------------------------------------------------------------------
typedef __attribute__((ext_vector_type(16))) __bf16 v16bf;
typedef __attribute__((ext_vector_type(8)))  float  v8f;

#define TILE_M 128
#define TILE_N 128
#define TILE_K 32
#define LDSS   40   // 32 + 8 pad (ushort units) -> 8B-aligned rows, bank decorrelated

__device__ __forceinline__ unsigned rnd_bf(float f) {        // round-to-nearest
  return __float_as_uint(f) + 0x8000u;
}

enum { EPI_NONE = 0, EPI_ADD = 1, EPI_SIG = 2, EPI_PRELU = 3, EPI_RES = 4 };

// ---------------------------------------------------------------------------
// staging helpers: global f32 -> registers -> bf16 in LDS
// ---------------------------------------------------------------------------
template <bool TRANSA>
__device__ __forceinline__ void load_regs(const float* __restrict__ Ab,
                                          const float* __restrict__ Bb,
                                          int lda, int ldb, int bm, int bn,
                                          int k0, int t,
                                          float4 ra[4], float4 rb[4])
{
  if (!TRANSA) {
    int r = t >> 3, c = (t & 7) * 4;
#pragma unroll
    for (int p = 0; p < 4; ++p)
      ra[p] = *(const float4*)(Ab + (long long)(bm + p * 32 + r) * lda + (k0 + c));
  } else {
    int kk = t >> 5, m4 = (t & 31) * 4;
#pragma unroll
    for (int p = 0; p < 4; ++p)
      ra[p] = *(const float4*)(Ab + (long long)(k0 + p * 8 + kk) * lda + (bm + m4));
  }
  int kk = t >> 5, n4 = (t & 31) * 4;
#pragma unroll
  for (int p = 0; p < 4; ++p)
    rb[p] = *(const float4*)(Bb + (long long)(k0 + p * 8 + kk) * ldb + (bn + n4));
}

template <bool TRANSA>
__device__ __forceinline__ void store_tile(unsigned short* __restrict__ As,
                                           unsigned short* __restrict__ Bs,
                                           int t, const float4 ra[4],
                                           const float4 rb[4])
{
  if (!TRANSA) {
    int r = t >> 3, c = (t & 7) * 4;
#pragma unroll
    for (int p = 0; p < 4; ++p) {
      unsigned u0 = rnd_bf(ra[p].x), u1 = rnd_bf(ra[p].y);
      unsigned u2 = rnd_bf(ra[p].z), u3 = rnd_bf(ra[p].w);
      uint2 pk;                              // v_perm_b32: pack two bf16/dword
      pk.x = __builtin_amdgcn_perm(u1, u0, 0x07060302u);
      pk.y = __builtin_amdgcn_perm(u3, u2, 0x07060302u);
      *(uint2*)&As[(p * 32 + r) * LDSS + c] = pk;
    }
  } else {
    int kk = t >> 5, m4 = (t & 31) * 4;
#pragma unroll
    for (int p = 0; p < 4; ++p) {
      int k = p * 8 + kk;
      As[(m4 + 0) * LDSS + k] = (unsigned short)(rnd_bf(ra[p].x) >> 16);
      As[(m4 + 1) * LDSS + k] = (unsigned short)(rnd_bf(ra[p].y) >> 16);
      As[(m4 + 2) * LDSS + k] = (unsigned short)(rnd_bf(ra[p].z) >> 16);
      As[(m4 + 3) * LDSS + k] = (unsigned short)(rnd_bf(ra[p].w) >> 16);
    }
  }
  int kk = t >> 5, n4 = (t & 31) * 4;
#pragma unroll
  for (int p = 0; p < 4; ++p) {
    int k = p * 8 + kk;
    Bs[(n4 + 0) * LDSS + k] = (unsigned short)(rnd_bf(rb[p].x) >> 16);
    Bs[(n4 + 1) * LDSS + k] = (unsigned short)(rnd_bf(rb[p].y) >> 16);
    Bs[(n4 + 2) * LDSS + k] = (unsigned short)(rnd_bf(rb[p].z) >> 16);
    Bs[(n4 + 3) * LDSS + k] = (unsigned short)(rnd_bf(rb[p].w) >> 16);
  }
}

__device__ __forceinline__ void compute_tile(const unsigned short* __restrict__ As,
                                             const unsigned short* __restrict__ Bs,
                                             int wm, int wn, int half, int idx,
                                             v8f acc[2][4])
{
  union FU { v16bf v; unsigned u[8]; };
  FU a[2], b[4];
#pragma unroll
  for (int mt = 0; mt < 2; ++mt) {
    const unsigned* rw = (const unsigned*)&As[(wm * 32 + mt * 16 + idx) * LDSS];
#pragma unroll
    for (int p = 0; p < 8; ++p)              // A: K = {0-7,16-23} / {8-15,24-31}
      a[mt].u[p] = rw[(p & 3) + half * 4 + ((p >> 2) << 3)];
  }
#pragma unroll
  for (int nt = 0; nt < 4; ++nt) {
    const unsigned* rw =
        (const unsigned*)&Bs[(wn * 64 + nt * 16 + idx) * LDSS];
#pragma unroll
    for (int p = 0; p < 8; ++p)              // B: K = {0-7,16-23} / {8-15,24-31}
      b[nt].u[p] = rw[(p & 3) + half * 4 + ((p >> 2) << 3)];
  }
#pragma unroll
  for (int mt = 0; mt < 2; ++mt)
#pragma unroll
    for (int nt = 0; nt < 4; ++nt)
      acc[mt][nt] = __builtin_amdgcn_wmma_f32_16x16x32_bf16(
          false, a[mt].v, false, b[nt].v, (short)0, acc[mt][nt], false, false);
}

// ---------------------------------------------------------------------------
// Tiled bf16 WMMA GEMM:  C[bz] = epi( A[bz](MxK) * B[bz](KxN) )
//  - software-pipelined: regs fetch tile k+1 while WMMA consumes tile k
//  - ping-pong LDS buffers -> single barrier per k-step
// ---------------------------------------------------------------------------
template <int EPI, bool TRANSA>
__global__ __launch_bounds__(256) void gemm_wmma(
    const float* __restrict__ A, long long sAb, int lda,
    const float* __restrict__ Bm, long long sBb, int ldb,
    float* __restrict__ C, long long sCb, int ldc,
    float* __restrict__ C2, int ldc2,
    const float* __restrict__ bias, float biasScale,
    const float* __restrict__ aux, long long sXb, int ldaux,
    const float* __restrict__ denom,
    const float* __restrict__ alphaPtr,
    int M, int N, int K)
{
  __shared__ __align__(16) unsigned short AsB[2][TILE_M * LDSS];
  __shared__ __align__(16) unsigned short BsB[2][TILE_N * LDSS];  // stored [n][k]

  const int t    = threadIdx.x;
  const int wave = t >> 5;
  const int lane = t & 31;
  const int half = lane >> 4;
  const int idx  = lane & 15;
  const int wm   = wave >> 1;   // 0..3 -> 32-row strip
  const int wn   = wave & 1;    // 0..1 -> 64-col strip

  const int bm = blockIdx.y * TILE_M;
  const int bn = blockIdx.x * TILE_N;
  const int bz = blockIdx.z;

  const float* Ab = A  + (long long)bz * sAb;
  const float* Bb = Bm + (long long)bz * sBb;

  v8f acc[2][4];
  v8f zero = {0.f, 0.f, 0.f, 0.f, 0.f, 0.f, 0.f, 0.f};
#pragma unroll
  for (int i = 0; i < 2; ++i)
#pragma unroll
    for (int j = 0; j < 4; ++j) acc[i][j] = zero;

  float4 ra[4], rb[4];
  load_regs<TRANSA>(Ab, Bb, lda, ldb, bm, bn, 0, t, ra, rb);
  store_tile<TRANSA>(AsB[0], BsB[0], t, ra, rb);
  if (TILE_K < K)
    load_regs<TRANSA>(Ab, Bb, lda, ldb, bm, bn, TILE_K, t, ra, rb);
  __syncthreads();

  int pb = 0;
  for (int k0 = 0; k0 < K; k0 += TILE_K, pb ^= 1) {
    compute_tile(AsB[pb], BsB[pb], wm, wn, half, idx, acc);
    if (k0 + TILE_K < K) {
      store_tile<TRANSA>(AsB[pb ^ 1], BsB[pb ^ 1], t, ra, rb);
      if (k0 + 2 * TILE_K < K) {
        load_regs<TRANSA>(Ab, Bb, lda, ldb, bm, bn, k0 + 2 * TILE_K, t, ra, rb);
        if (k0 + 3 * TILE_K < K) {   // L2 prefetch two tiles ahead
          __builtin_prefetch(Ab + (TRANSA ? (long long)(k0 + 3 * TILE_K) * lda + bm
                                          : (long long)bm * lda + (k0 + 3 * TILE_K)),
                             0, 1);
          __builtin_prefetch(Bb + (long long)(k0 + 3 * TILE_K) * ldb + bn, 0, 1);
        }
      }
      __syncthreads();
    }
  }

  const float alpha = (EPI == EPI_PRELU) ? *alphaPtr : 0.f;

#pragma unroll
  for (int mt = 0; mt < 2; ++mt) {
#pragma unroll
    for (int nt = 0; nt < 4; ++nt) {
#pragma unroll
      for (int e = 0; e < 8; ++e) {
        int row = bm + wm * 32 + mt * 16 + e + half * 8;
        int col = bn + wn * 64 + nt * 16 + idx;
        long long fr = (long long)bz * M + row;
        float v = acc[mt][nt][e];
        if (EPI == EPI_ADD) {
          v += aux[(long long)bz * sXb + (long long)row * ldaux + col];
        } else if (EPI == EPI_SIG) {
          v += bias[col] * biasScale;
          v = 1.0f / (1.0f + __expf(-v));
        } else if (EPI == EPI_PRELU) {
          v += aux[fr * ldaux + col] + bias[col] * biasScale;
          v /= denom[fr];
          v = (v >= 0.f) ? v : alpha * v;
        } else if (EPI == EPI_RES) {
          v += bias[col] * biasScale + aux[fr * ldaux + col];
        }
        C[(long long)bz * sCb + (long long)row * ldc + col] = v;
        if (EPI == EPI_PRELU && C2) C2[fr * ldc2 + col] = v;
      }
    }
  }
}

// ---------------------------------------------------------------------------
// Once-per-call prep: denom, frontadj, backadj, domain/redomain gather index
// ---------------------------------------------------------------------------
__global__ void prep_kernel(const float* __restrict__ adj,
                            const float* __restrict__ depmap,
                            const float* __restrict__ domain_id,
                            const float* __restrict__ redomain_id,
                            float* denom, float* frontadj, float* backadj,
                            int* idxF, int* idxR)
{
  int r = blockIdx.x * blockDim.x + threadIdx.x;
  if (r >= B_ * S_) return;
  int b = r >> 9, s = r & (S_ - 1);
  const float* arow = adj + (long long)r * S_;
  const float* drow = depmap + (long long)r * S_;
  const float* acol = adj + (long long)b * SS + s;
  float den = 0.f, fa = 0.f, ba = 0.f;
  for (int j = 0; j < S_; ++j) {
    float a = arow[j], dm = drow[j];
    den += a;
    fa  += a * dm;
    ba  += acol[(long long)j * S_] * dm;
  }
  denom[r] = den + 1.f;
  frontadj[r] = fa;
  backadj[r] = ba;

  float sum = 0.f; int id = -1;
  for (int k = 0; k < DOM_; ++k) {
    float v = domain_id[(long long)r * DOM_ + k];
    sum += v; if (v > 0.5f) id = k;
  }
  idxF[r] = (sum > 0.5f) ? id : -1;
  sum = 0.f; id = -1;
  for (int k = 0; k < DOM_; ++k) {
    float v = redomain_id[(long long)r * DOM_ + k];
    sum += v; if (v > 0.5f) id = k;
  }
  idxR[r] = (sum > 0.5f) ? id : -1;
}

// pooled[b,dom,d] = max_s ( domain[b,s,dom] ? -1e4 : outputs[b,s,d] )
__global__ void pool_kernel(const float* __restrict__ outputs,
                            const unsigned char* __restrict__ domain,
                            float* __restrict__ pooled)
{
  int i = blockIdx.x * blockDim.x + threadIdx.x;
  if (i >= B_ * DOM_ * D_) return;
  int d = i & (D_ - 1);
  int dom = (i / D_) & (DOM_ - 1);
  int b = i / (D_ * DOM_);
  const float* ob = outputs + (long long)b * SD + d;
  const unsigned char* db = domain + (long long)b * S_ * DOM_ + dom;
  float m = -3.4e38f;
  for (int s = 0; s < S_; ++s) {
    float v = db[(long long)s * DOM_] ? -10000.f : ob[(long long)s * D_];
    m = fmaxf(m, v);
  }
  pooled[i] = m;
}

// build frontoutputs/backoutputs directly inside the concat buffers
__global__ void ew_frontback(const float* __restrict__ outputs,
                             const float* __restrict__ Y1,
                             const float* __restrict__ dg,
                             const float* __restrict__ frontrel,
                             const float* __restrict__ backrel,
                             const int* __restrict__ idxF,
                             const int* __restrict__ idxR,
                             float* __restrict__ concatF,
                             float* __restrict__ concatB)
{
  long long i = (long long)blockIdx.x * blockDim.x + threadIdx.x;
  if (i >= BSD) return;
  int d = (int)(i & (D_ - 1));
  long long r = i >> 9;
  int b = (int)(r >> 9);
  int iF = idxF[r], iR = idxR[r];
  float fo = (iF >= 0) ? dg[((long long)b * DOM_ + iF) * D_ + d] : outputs[i];
  float bo = (iR >= 0) ? dg[((long long)b * DOM_ + iR) * D_ + d] : Y1[i];
  concatF[r * (2 * D_) + d]      = fo;
  concatF[r * (2 * D_) + D_ + d] = frontrel[i];
  concatB[r * (2 * D_) + d]      = bo;
  concatB[r * (2 * D_) + D_ + d] = backrel[i];
}

// Xf = frontadj*frontdep*frontout ; backterm = backadj*backdep*backout ;
// BxRel = outputs .* rel
__global__ void ew_x(const float* __restrict__ concatF,
                     const float* __restrict__ concatB,
                     const float* __restrict__ frontdep,
                     const float* __restrict__ backdep,
                     const float* __restrict__ frontadj,
                     const float* __restrict__ backadj,
                     const float* __restrict__ outputs,
                     const float* __restrict__ rel,
                     float* __restrict__ Xf,
                     float* __restrict__ backterm,
                     float* __restrict__ BxRel)
{
  long long i = (long long)blockIdx.x * blockDim.x + threadIdx.x;
  if (i >= BSD) return;
  int d = (int)(i & (D_ - 1));
  long long r = i >> 9;
  float fo = concatF[r * (2 * D_) + d];
  float bo = concatB[r * (2 * D_) + d];
  Xf[i]       = frontadj[r] * frontdep[i] * fo;
  backterm[i] = backadj[r]  * backdep[i]  * bo;
  BxRel[i]    = outputs[i] * rel[i];
}

// ---------------------------------------------------------------------------
// Host launch
// ---------------------------------------------------------------------------
extern "C" void kernel_launch(void* const* d_in, const int* in_sizes, int n_in,
                              void* d_out, int out_size, void* d_ws, size_t ws_size,
                              hipStream_t stream)
{
  const float* adj         = (const float*)d_in[0];
  const unsigned char* dom = (const unsigned char*)d_in[1];   // bool [B,S,DOM,1]
  const float* domain_id   = (const float*)d_in[2];
  const float* redomain_id = (const float*)d_in[3];
  const float* frontrel    = (const float*)d_in[4];
  const float* backrel     = (const float*)d_in[5];
  const float* depmap      = (const float*)d_in[6];
  const float* rel         = (const float*)d_in[7];
  const float* gcn         = (const float*)d_in[8];
  // d_in[9] = mask (unused by forward)
  const float* W_layers    = (const float*)d_in[10];
  const float* b_layers    = (const float*)d_in[11];
  const float* prelu_a     = (const float*)d_in[12];
  const float* W_dg        = (const float*)d_in[13];
  const float* b_dg        = (const float*)d_in[14];
  const float* W_mg        = (const float*)d_in[15];
  const float* b_mg        = (const float*)d_in[16];
  const float* W_out       = (const float*)d_in[17];
  const float* b_out       = (const float*)d_in[18];

  // workspace layout (with reuse: Y1->Xf, concatF->AxpO, concatB[0:BSD]->delta)
  float* w = (float*)d_ws;
  size_t o = 0;
  auto alloc = [&](long long n) { float* p = w + o; o += (size_t)n; return p; };
  float* denomv    = alloc(B_ * S_);
  float* frontadjv = alloc(B_ * S_);
  float* backadjv  = alloc(B_ * S_);
  int*   idxF      = (int*)alloc(B_ * S_);
  int*   idxR      = (int*)alloc(B_ * S_);
  float* pooled    = alloc((long long)B_ * DOM_ * D_);
  float* dg        = alloc((long long)B_ * DOM_ * D_);
  float* buf1      = alloc(BSD);        // Y1, then Xf
  float* concatF   = alloc(2 * BSD);    // later AxpO
  float* concatB   = alloc(2 * BSD);    // first BSD later delta
  float* frontdep  = alloc(BSD);
  float* backdep   = alloc(BSD);
  float* backterm  = alloc(BSD);
  float* BxRel     = alloc(BSD);
  float* outbuf    = alloc(BSD);
  float* outsCat   = alloc(3 * BSD);
  float* delta = concatB;
  float* AxpO  = concatF;

  const int ewBlocks = (int)(BSD / 256);

  prep_kernel<<<(B_ * S_ + 255) / 256, 256, 0, stream>>>(
      adj, depmap, domain_id, redomain_id, denomv, frontadjv, backadjv, idxF, idxR);

  for (int l = 0; l < L_; ++l) {
    const float* cur = (l == 0) ? gcn : outbuf;

    // domain max-pool + gate: dg = sigmoid(pooled @ W_dg + b_dg)
    pool_kernel<<<(B_ * DOM_ * D_ + 255) / 256, 256, 0, stream>>>(cur, dom, pooled);
    gemm_wmma<EPI_SIG, false><<<dim3(4, 1, 1), 256, 0, stream>>>(
        pooled, 0, D_, W_dg, 0, D_, dg, 0, D_, nullptr, 0,
        b_dg, 1.0f, nullptr, 0, 0, nullptr, nullptr, B_ * DOM_, D_, D_);

    // Y1 = rel @ outputs  (batched)
    gemm_wmma<EPI_NONE, false><<<dim3(4, 4, B_), 256, 0, stream>>>(
        rel, SS, S_, cur, SD, D_, buf1, SD, D_, nullptr, 0,
        nullptr, 0.f, nullptr, 0, 0, nullptr, nullptr, S_, D_, S_);

    // front/back outputs + concat buffers
    ew_frontback<<<ewBlocks, 256, 0, stream>>>(
        cur, buf1, dg, frontrel, backrel, idxF, idxR, concatF, concatB);

    // frontdepadj / backdepadj = sigmoid([x, rel_feat] @ W_mg + b_mg)
    gemm_wmma<EPI_SIG, false><<<dim3(4, 64, 1), 256, 0, stream>>>(
        concatF, 0, 2 * D_, W_mg, 0, D_, frontdep, 0, D_, nullptr, 0,
        b_mg, 1.0f, nullptr, 0, 0, nullptr, nullptr, B_ * S_, D_, 2 * D_);
    gemm_wmma<EPI_SIG, false><<<dim3(4, 64, 1), 256, 0, stream>>>(
        concatB, 0, 2 * D_, W_mg, 0, D_, backdep, 0, D_, nullptr, 0,
        b_mg, 1.0f, nullptr, 0, 0, nullptr, nullptr, B_ * S_, D_, 2 * D_);

    // elementwise gating products
    ew_x<<<ewBlocks, 256, 0, stream>>>(
        concatF, concatB, frontdep, backdep, frontadjv, backadjv, cur, rel,
        buf1 /*Xf*/, backterm, BxRel);

    // delta = rel^T @ Xf + backterm  (batched, transposed A, fused add)
    gemm_wmma<EPI_ADD, true><<<dim3(4, 4, B_), 256, 0, stream>>>(
        rel, SS, S_, buf1, SD, D_, delta, SD, D_, nullptr, 0,
        nullptr, 0.f, backterm, SD, D_, nullptr, nullptr, S_, D_, S_);

    // AxpO = adj @ (outputs .* rel) + outputs  (batched, fused add)
    gemm_wmma<EPI_ADD, false><<<dim3(4, 4, B_), 256, 0, stream>>>(
        adj, SS, S_, BxRel, SD, D_, AxpO, SD, D_, nullptr, 0,
        nullptr, 0.f, cur, SD, D_, nullptr, nullptr, S_, D_, S_);

    // outputs = PReLU( (AxpO @ W_l + 2*b_l + delta) / denom ), dual-store
    gemm_wmma<EPI_PRELU, false><<<dim3(4, 64, 1), 256, 0, stream>>>(
        AxpO, 0, D_, W_layers + (long long)l * D_ * D_, 0, D_,
        outbuf, 0, D_, outsCat + (long long)l * D_, 3 * D_,
        b_layers + (long long)l * D_, 2.0f, delta, 0, D_,
        denomv, prelu_a + l, B_ * S_, D_, D_);
  }

  // out = outsCat @ W_out + b_out + gcn_inputs
  gemm_wmma<EPI_RES, false><<<dim3(4, 64, 1), 256, 0, stream>>>(
      outsCat, 0, 3 * D_, W_out, 0, D_, (float*)d_out, 0, D_, nullptr, 0,
      b_out, 1.0f, gcn, 0, D_, nullptr, nullptr, B_ * S_, D_, 3 * D_);
}
